// EnergyDipolesMACE_43576738185437
// MI455X (gfx1250) — matrix operand order repaired and checked by reading, not compiled, not run on hardware
//
#include <hip/hip_runtime.h>
#include <math.h>

typedef unsigned short u16b;
typedef __attribute__((ext_vector_type(16))) __bf16 v16bf;
typedef __attribute__((ext_vector_type(8)))  float  v8f;

#define F_DIM 128
#define G_DIM 8
#define Z_DIM 10
#define NBES 8
#define RMAXF 5.0f
#define AVGF 16.0f
#define PI_F 3.14159265358979323846f

__constant__ int LMAP_c[16] = {0,1,1,1,2,2,2,2,2,3,3,3,3,3,3,3};
static const int LMAP_h[16] = {0,1,1,1,2,2,2,2,2,3,3,3,3,3,3,3};

// ---------------- helpers ----------------
__device__ __forceinline__ u16b f2bf(float f) {
    union { float f; unsigned u; } v; v.f = f;
    unsigned u = v.u;
    unsigned lsb = (u >> 16) & 1u;
    u += 0x7fffu + lsb;
    return (u16b)(u >> 16);
}
__device__ __forceinline__ float bf2f(u16b h) {
    union { unsigned u; float f; } v; v.u = ((unsigned)h) << 16; return v.f;
}
__device__ __forceinline__ float silu_f(float x) {
    return x / (1.f + __expf(-x));
}
__device__ __forceinline__ float silu_g(float x) {   // d/dx silu(x)
    float s = 1.f / (1.f + __expf(-x));
    return s * (1.f + x * (1.f - s));
}

// ---------------- utility kernels ----------------
__global__ void k_zero_f32(float* p, long n) {
    long i = (long)blockIdx.x * blockDim.x + threadIdx.x;
    long stride = (long)gridDim.x * blockDim.x;
    for (; i < n; i += stride) p[i] = 0.f;
}
__global__ void k_f2bf_arr(const float* __restrict__ src, u16b* __restrict__ dst, long n) {
    long i = (long)blockIdx.x * blockDim.x + threadIdx.x;
    long stride = (long)gridDim.x * blockDim.x;
    for (; i < n; i += stride) dst[i] = f2bf(src[i]);
}
__global__ void k_f2bf_scale(const float* __restrict__ src, u16b* __restrict__ dst, long n, float scale) {
    long i = (long)blockIdx.x * blockDim.x + threadIdx.x;
    long stride = (long)gridDim.x * blockDim.x;
    for (; i < n; i += stride) dst[i] = f2bf(src[i] * scale);
}

// ---------------- weight packer -> WMMA B-fragment order ----------------
// dst[((kb*tilesN + tj)*32 + lane)*16 + e]  where k = kb*32 + (lane>>4)*16 + e, n = tj*16 + (lane&15)
__global__ void k_pack_b(const float* __restrict__ W, u16b* __restrict__ dst,
                         int Kvalid, int Nc, int rs, int cs, int kblocks, int tilesN)
{
    int idx = blockIdx.x * blockDim.x + threadIdx.x;
    int total = kblocks * tilesN * 512;
    if (idx >= total) return;
    int e    = idx & 15;
    int lane = (idx >> 4) & 31;
    int blk  = idx >> 9;
    int tj   = blk % tilesN;
    int kb   = blk / tilesN;
    int ncol = tj * 16 + (lane & 15);
    int k    = kb * 32 + (lane >> 4) * 16 + e;
    u16b v = 0;
    if (k < Kvalid && ncol < Nc) v = f2bf(W[(size_t)k * rs + (size_t)ncol * cs]);
    dst[idx] = v;
}

// ---------------- WMMA GEMM: NT output tiles per wave (A-fragment reuse) ----------------
// requires M%16==0, K%32==0, lda%8==0, (Nc+15)/16 % NT == 0, B pre-packed
template<int NT>
__global__ void k_gemm_bf16_t(const u16b* __restrict__ A, const u16b* __restrict__ Bp,
                              float* __restrict__ C, u16b* __restrict__ Cbf,
                              u16b* __restrict__ Zpre, const u16b* __restrict__ Zin,
                              int M, int K, int Nc, int lda, int ldc,
                              float alpha, int act, int accum)
{
    int wave = (blockIdx.x * blockDim.x + threadIdx.x) >> 5;
    int lane = threadIdx.x & 31;
    int tilesN = (Nc + 15) >> 4;
    int groupsN = tilesN / NT;
    int tilesM = M >> 4;
    if (wave >= tilesM * groupsN) return;
    int ti = wave / groupsN, tg = wave - ti * groupsN;
    int row0 = ti << 4;
    int m = lane & 15, half = lane >> 4;

    union Frag { v16bf v; uint4 q[2]; };
    Frag af;
    v8f acc[NT];
    #pragma unroll
    for (int t = 0; t < NT; ++t) acc[t] = (v8f){};
    const u16b* Ar = A + (size_t)(row0 + m) * lda + half * 16;
    const u16b* Br = Bp + (((size_t)tg * NT) * 32 + lane) * 16;
    size_t bstep = (size_t)tilesN * 512;
    int kblocks = K >> 5;
    for (int kb = 0; kb < kblocks; ++kb) {
        af.q[0] = *(const uint4*)(Ar);
        af.q[1] = *(const uint4*)(Ar + 8);
        #pragma unroll
        for (int t = 0; t < NT; ++t) {
            Frag bfm;
            bfm.q[0] = *(const uint4*)(Br + t * 512);
            bfm.q[1] = *(const uint4*)(Br + t * 512 + 8);
            acc[t] = __builtin_amdgcn_wmma_f32_16x16x32_bf16(false, af.v, false, bfm.v,
                                                             (short)0, acc[t], false, false);
        }
        Ar += 32;
        Br += bstep;
    }
    #pragma unroll
    for (int t = 0; t < NT; ++t) {
        int col = (tg * NT + t) * 16 + m;
        if (col < Nc) {
            #pragma unroll
            for (int j = 0; j < 8; ++j) {
                int row = row0 + j + half * 8;
                size_t idx = (size_t)row * ldc + col;
                float x = alpha * acc[t][j];
                if (accum) x += C[idx];
                if (act == 1) {
                    if (Zpre) Zpre[idx] = f2bf(x);
                    x = silu_f(x);
                } else if (act == 2) {
                    x *= silu_g(bf2f(Zin[idx]));
                }
                if (C)   C[idx] = x;
                if (Cbf) Cbf[idx] = f2bf(x);
            }
        }
    }
}

// ---------------- edge geometry ----------------
__global__ void k_edges(const float* __restrict__ pos, const float* __restrict__ shifts,
                        const int* __restrict__ snd, const int* __restrict__ rcv,
                        float* __restrict__ rbuf, float* __restrict__ ubuf,
                        float* __restrict__ Y, u16b* __restrict__ efbf, int E)
{
    int e = blockIdx.x * blockDim.x + threadIdx.x;
    if (e >= E) return;
    int s = snd[e], d = rcv[e];
    float vx = pos[d * 3 + 0] - pos[s * 3 + 0] + shifts[e * 3 + 0];
    float vy = pos[d * 3 + 1] - pos[s * 3 + 1] + shifts[e * 3 + 1];
    float vz = pos[d * 3 + 2] - pos[s * 3 + 2] + shifts[e * 3 + 2];
    float rr = sqrtf(vx * vx + vy * vy + vz * vz);
    rr = fmaxf(rr, 1e-9f);
    float x = vx / rr, y = vy / rr, z = vz / rr;
    rbuf[e] = rr;
    ubuf[e * 3 + 0] = x; ubuf[e * 3 + 1] = y; ubuf[e * 3 + 2] = z;

    const float s3  = 1.7320508075688772f;
    const float s5  = 2.2360679774997896f;
    const float s15 = 3.872983346207417f;
    const float c9  = 2.0916500663351889f;
    const float c10 = 10.246950765959598f;
    const float c11 = 1.6201851746019651f;
    const float c12 = 1.3228756555322954f;
    const float c14 = 5.123475382979799f;

    float* Ye = Y + (size_t)e * 16;
    Ye[0]  = 1.f;
    Ye[1]  = s3 * x;  Ye[2] = s3 * y;  Ye[3] = s3 * z;
    Ye[4]  = s15 * x * y;
    Ye[5]  = s15 * y * z;
    Ye[6]  = 0.5f * s5 * (3.f * z * z - 1.f);
    Ye[7]  = s15 * x * z;
    Ye[8]  = 0.5f * s15 * (x * x - y * y);
    Ye[9]  = c9 * y * (3.f * x * x - y * y);
    Ye[10] = c10 * x * y * z;
    Ye[11] = c11 * y * (5.f * z * z - 1.f);
    Ye[12] = c12 * z * (5.f * z * z - 3.f);
    Ye[13] = c11 * x * (5.f * z * z - 1.f);
    Ye[14] = c14 * z * (x * x - y * y);
    Ye[15] = c9 * x * (x * x - 3.f * y * y);

    float u = rr / RMAXF;
    float cut = 0.f;
    if (u < 1.f) {
        float u2 = u * u;
        float u5 = u2 * u2 * u;
        cut = 1.f - 21.f * u5 + 35.f * u5 * u - 15.f * u5 * u2;
    }
    float pref = sqrtf(2.f / RMAXF) / rr * cut;
    u16b* row = efbf + (size_t)e * 32;
    #pragma unroll
    for (int n = 1; n <= NBES; ++n)
        row[n - 1] = f2bf(pref * __sinf((float)n * PI_F * rr / RMAXF));
    #pragma unroll
    for (int n = NBES; n < 32; ++n) row[n] = 0;
}

// ---------------- node embedding ----------------
__global__ void k_embed(const float* __restrict__ attrs, const float* __restrict__ ae,
                        const float* __restrict__ Wemb, const int* __restrict__ batch,
                        int* __restrict__ typ, u16b* __restrict__ hbf,
                        float* __restrict__ ne0, float* __restrict__ te, int N)
{
    int n = blockIdx.x;
    int c = threadIdx.x;
    if (n >= N) return;
    __shared__ int st;
    if (c == 0) {
        int t = 0; float best = attrs[(size_t)n * Z_DIM];
        for (int zc = 1; zc < Z_DIM; ++zc) {
            float v = attrs[(size_t)n * Z_DIM + zc];
            if (v > best) { best = v; t = zc; }
        }
        st = t; typ[n] = t;
        float e = ae[t];
        ne0[n] = e;
        atomicAdd(&te[batch[n]], e);
    }
    __syncthreads();
    hbf[(size_t)n * F_DIM + c] = f2bf(Wemb[(size_t)st * F_DIM + c]);
}

// ---------------- message scatters ----------------
__global__ void k_msg1(const float* __restrict__ Y, const float* __restrict__ R,
                       const u16b* __restrict__ hs,
                       const int* __restrict__ snd, const int* __restrict__ rcv,
                       float* __restrict__ A)
{
    int e = blockIdx.x;
    int c = threadIdx.x;
    int s = snd[e], d = rcv[e];
    float h = bf2f(hs[(size_t)s * F_DIM + c]);
    const float* Ye = Y + (size_t)e * 16;
    const float* Re = R + (size_t)e * 4;
    float* Ad = A + (size_t)d * 2048 + c;
    #pragma unroll
    for (int k = 0; k < 16; ++k)
        atomicAdd(Ad + k * F_DIM, Re[LMAP_c[k]] * Ye[k] * h);
}
__global__ void k_msg2(const float* __restrict__ Y, const float* __restrict__ R2,
                       const u16b* __restrict__ hs0, const u16b* __restrict__ hv,
                       const int* __restrict__ snd, const int* __restrict__ rcv,
                       float* __restrict__ A)
{
    int e = blockIdx.x;
    int c = threadIdx.x;
    int s = snd[e], d = rcv[e];
    const float* Ye = Y + (size_t)e * 16;
    const float* Re = R2 + (size_t)e * 5;
    float h = bf2f(hs0[(size_t)s * F_DIM + c]);
    float* Ad = A + (size_t)d * 2048 + c;
    float extra = Re[4] * (Ye[1] * bf2f(hv[((size_t)s * 3 + 0) * F_DIM + c])
                         + Ye[2] * bf2f(hv[((size_t)s * 3 + 1) * F_DIM + c])
                         + Ye[3] * bf2f(hv[((size_t)s * 3 + 2) * F_DIM + c]));
    atomicAdd(Ad, Re[0] * Ye[0] * h + extra);
    #pragma unroll
    for (int k = 1; k < 16; ++k)
        atomicAdd(Ad + k * F_DIM, Re[LMAP_c[k]] * Ye[k] * h);
}

// ---------------- nonlinearity fwd ----------------
__global__ void k_nonlin(const float* __restrict__ Amix,
                         const float* __restrict__ wp0, const float* __restrict__ wp1,
                         u16b* __restrict__ h0in, u16b* __restrict__ h1in)
{
    int n = blockIdx.x;
    int c = threadIdx.x;
    const float* a = Amix + (size_t)n * 2048;
    float a0 = a[c];
    float S = 0.f;
    #pragma unroll
    for (int k = 0; k < 16; ++k) { float v = a[k * F_DIM + c]; S += v * v; }
    h0in[(size_t)n * F_DIM + c] = f2bf(a0 + wp0[c] * S);
    float p = wp1[c] * a0;
    #pragma unroll
    for (int mm = 0; mm < 3; ++mm) {
        float v = a[(1 + mm) * F_DIM + c];
        h1in[((size_t)n * 3 + mm) * F_DIM + c] = f2bf(v + p * v);
    }
}

// ---------------- nonlinearity bwd ----------------
__global__ void k_nonlin_bwd(const float* __restrict__ Amix,
                             const float* __restrict__ g0, const float* __restrict__ g1,
                             const float* __restrict__ wp0, const float* __restrict__ wp1,
                             float* __restrict__ gA)
{
    int n = blockIdx.x;
    int c = threadIdx.x;
    const float* a = Amix + (size_t)n * 2048;
    float* g = gA + (size_t)n * 2048;
    float gh0 = g0[(size_t)n * F_DIM + c];
    float a0 = a[c];
    float w0 = wp0[c], w1 = wp1[c];
    float gk[16];
    #pragma unroll
    for (int k = 0; k < 16; ++k) gk[k] = gh0 * 2.f * w0 * a[k * F_DIM + c];
    gk[0] += gh0;
    if (g1) {
        #pragma unroll
        for (int mm = 0; mm < 3; ++mm) {
            float g1m = g1[((size_t)n * 3 + mm) * F_DIM + c];
            float am  = a[(1 + mm) * F_DIM + c];
            gk[1 + mm] += g1m * (1.f + w1 * a0);
            gk[0]      += w1 * g1m * am;
        }
    }
    #pragma unroll
    for (int k = 0; k < 16; ++k) g[k * F_DIM + c] = gk[k];
}

// ---------------- e1 / d1 ----------------
__global__ void k_e1d1(const float* __restrict__ h0f, const float* __restrict__ h1f,
                       const float* __restrict__ we1, const float* __restrict__ wd1,
                       const int* __restrict__ batch, float* __restrict__ te,
                       float* __restrict__ d1, int N)
{
    int n = blockIdx.x * blockDim.x + threadIdx.x;
    if (n >= N) return;
    float e = 0.f;
    for (int c = 0; c < F_DIM; ++c) e += h0f[(size_t)n * F_DIM + c] * we1[c];
    atomicAdd(&te[batch[n]], e);
    for (int mm = 0; mm < 3; ++mm) {
        float dd = 0.f;
        const float* row = h1f + ((size_t)n * 3 + mm) * F_DIM;
        for (int c = 0; c < F_DIM; ++c) dd += row[c] * wd1[c];
        d1[(size_t)n * 3 + mm] = dd;
    }
}

// ---------------- species residual fwd ----------------
__global__ void k_sc_add(const float* __restrict__ h0f, const float* __restrict__ h1f,
                         const int* __restrict__ typ, const float* __restrict__ Wsc2,
                         float* __restrict__ B0f, float* __restrict__ B1f,
                         u16b* __restrict__ B0bf, u16b* __restrict__ B1bf)
{
    __shared__ float sh0[F_DIM];
    __shared__ float sh1[3][F_DIM];
    int n = blockIdx.x;
    int d = threadIdx.x;
    sh0[d] = h0f[(size_t)n * F_DIM + d];
    for (int mm = 0; mm < 3; ++mm) sh1[mm][d] = h1f[((size_t)n * 3 + mm) * F_DIM + d];
    __syncthreads();
    int t = typ[n];
    const float* W0 = Wsc2 + (size_t)t * 2 * F_DIM * F_DIM;
    const float* W1 = W0 + F_DIM * F_DIM;
    float a0 = 0.f, a1 = 0.f, a2 = 0.f, a3 = 0.f;
    for (int c = 0; c < F_DIM; ++c) {
        float w0 = W0[(size_t)c * F_DIM + d];
        float w1 = W1[(size_t)c * F_DIM + d];
        a0 += sh0[c] * w0;
        a1 += sh1[0][c] * w1;
        a2 += sh1[1][c] * w1;
        a3 += sh1[2][c] * w1;
    }
    size_t i0 = (size_t)n * F_DIM + d;
    float b0 = B0f[i0] + a0;
    B0f[i0] = b0; B0bf[i0] = f2bf(b0);
    size_t i1 = ((size_t)n * 3 + 0) * F_DIM + d; float b;
    b = B1f[i1] + a1; B1f[i1] = b; B1bf[i1] = f2bf(b);
    i1 = ((size_t)n * 3 + 1) * F_DIM + d;
    b = B1f[i1] + a2; B1f[i1] = b; B1bf[i1] = f2bf(b);
    i1 = ((size_t)n * 3 + 2) * F_DIM + d;
    b = B1f[i1] + a3; B1f[i1] = b; B1bf[i1] = f2bf(b);
}

// ---------------- species residual bwd ----------------
__global__ void k_sc_bwd(const float* __restrict__ gB0f, const int* __restrict__ typ,
                         const float* __restrict__ Wsc2, const float* __restrict__ we1,
                         float* __restrict__ gh0f)
{
    __shared__ float sg[F_DIM];
    int n = blockIdx.x;
    int c = threadIdx.x;
    sg[c] = gB0f[(size_t)n * F_DIM + c];
    __syncthreads();
    int t = typ[n];
    const float* W0 = Wsc2 + (size_t)t * 2 * F_DIM * F_DIM;   // [c][d]
    float acc = we1[c];
    const float* row = W0 + (size_t)c * F_DIM;
    for (int d = 0; d < F_DIM; ++d) acc += sg[d] * row[d];
    gh0f[(size_t)n * F_DIM + c] = acc;
}

// ---------------- readout ----------------
__global__ void k_readout(const float* __restrict__ sbuf, const float* __restrict__ vbuf,
                          const float* __restrict__ d1, const float* __restrict__ we2,
                          const float* __restrict__ wd2, const int* __restrict__ batch,
                          const float* __restrict__ charges, const float* __restrict__ pos,
                          float* __restrict__ te, float* __restrict__ td,
                          float* __restrict__ ad, int N)
{
    int n = blockIdx.x * blockDim.x + threadIdx.x;
    if (n >= N) return;
    const float* s = sbuf + (size_t)n * 32;
    float gate[16];
    float e2 = 0.f;
    #pragma unroll
    for (int k = 0; k < 16; ++k) {
        e2 += silu_f(s[k]) * we2[k];
        gate[k] = silu_f(s[16 + k]);
    }
    int g = batch[n];
    atomicAdd(&te[g], e2);
    float q = charges[n];
    for (int mm = 0; mm < 3; ++mm) {
        float d2 = 0.f;
        const float* v = vbuf + ((size_t)n * 3 + mm) * 16;
        #pragma unroll
        for (int k = 0; k < 16; ++k) d2 += v[k] * gate[k] * wd2[k];
        float a = d1[(size_t)n * 3 + mm] + d2;
        ad[(size_t)n * 3 + mm] = a;
        atomicAdd(&td[g * 3 + mm], a + q * pos[(size_t)n * 3 + mm]);
    }
}

// ---------------- ds = we2 * silu'(s[:16]) ----------------
__global__ void k_ds(const float* __restrict__ sbuf, const float* __restrict__ we2,
                     u16b* __restrict__ dsbf, int N)
{
    int idx = blockIdx.x * blockDim.x + threadIdx.x;
    if (idx >= N * 32) return;
    int j = idx & 31;
    float v = 0.f;
    if (j < 16) v = we2[j] * silu_g(sbuf[idx]);
    dsbf[idx] = f2bf(v);
}

// ---------------- edge backward (layer 2) ----------------
#define RED_PAD 132
__global__ void k_edge_bwd2(const float* __restrict__ Y, const float* __restrict__ R2,
                            const u16b* __restrict__ hs0bf, const u16b* __restrict__ hvbf,
                            const float* __restrict__ gA2,
                            const int* __restrict__ snd, const int* __restrict__ rcv,
                            float* __restrict__ gY, u16b* __restrict__ gR2bf,
                            float* __restrict__ ghs0f, float* __restrict__ ghvf)
{
    __shared__ float part[19][RED_PAD];
    __shared__ float dots[19];
    __shared__ float grloc[5];
    int e = blockIdx.x;
    int c = threadIdx.x;
    int s = snd[e], d = rcv[e];
    float hs0c = bf2f(hs0bf[(size_t)s * F_DIM + c]);
    float hv0 = bf2f(hvbf[((size_t)s * 3 + 0) * F_DIM + c]);
    float hv1 = bf2f(hvbf[((size_t)s * 3 + 1) * F_DIM + c]);
    float hv2 = bf2f(hvbf[((size_t)s * 3 + 2) * F_DIM + c]);
    float ga[16];
    const float* gar = gA2 + (size_t)d * 2048 + c;
    #pragma unroll
    for (int k = 0; k < 16; ++k) ga[k] = gar[k * F_DIM];
    #pragma unroll
    for (int k = 0; k < 16; ++k) part[k][c] = ga[k] * hs0c;
    part[16][c] = ga[0] * hv0;
    part[17][c] = ga[0] * hv1;
    part[18][c] = ga[0] * hv2;
    __syncthreads();
    if (c < 19) {
        float sacc = 0.f;
        for (int i = 0; i < F_DIM; ++i) sacc += part[c][i];
        dots[c] = sacc;
    }
    __syncthreads();
    const float* Ye = Y + (size_t)e * 16;
    const float* Re = R2 + (size_t)e * 5;
    const float inv = 1.f / AVGF;
    if (c < 16) {
        float v = Re[LMAP_c[c]] * dots[c];
        if (c >= 1 && c <= 3) v += Re[4] * dots[16 + c - 1];
        gY[(size_t)e * 16 + c] = v * inv;
    }
    if (c < 5) {
        float v = 0.f;
        if (c < 4) {
            for (int k = 0; k < 16; ++k) if (LMAP_c[k] == c) v += Ye[k] * dots[k];
        } else {
            v = Ye[1] * dots[16] + Ye[2] * dots[17] + Ye[3] * dots[18];
        }
        grloc[c] = v * inv;
    }
    __syncthreads();
    if (c < 32) gR2bf[(size_t)e * 32 + c] = (c < 5) ? f2bf(grloc[c]) : (u16b)0;
    float gv = 0.f;
    #pragma unroll
    for (int k = 0; k < 16; ++k) gv += Re[LMAP_c[k]] * Ye[k] * ga[k];
    atomicAdd(&ghs0f[(size_t)s * F_DIM + c], gv * inv);
    float g0 = Re[4] * ga[0] * inv;
    atomicAdd(&ghvf[((size_t)s * 3 + 0) * F_DIM + c], g0 * Ye[1]);
    atomicAdd(&ghvf[((size_t)s * 3 + 1) * F_DIM + c], g0 * Ye[2]);
    atomicAdd(&ghvf[((size_t)s * 3 + 2) * F_DIM + c], g0 * Ye[3]);
}

// ---------------- edge backward (layer 1) ----------------
__global__ void k_edge_bwd1(const float* __restrict__ Y, const float* __restrict__ R1,
                            const u16b* __restrict__ hsbf, const float* __restrict__ gA1,
                            const int* __restrict__ snd, const int* __restrict__ rcv,
                            float* __restrict__ gY, u16b* __restrict__ gR1bf)
{
    __shared__ float part[16][RED_PAD];
    __shared__ float dots[16];
    __shared__ float grloc[4];
    int e = blockIdx.x;
    int c = threadIdx.x;
    int s = snd[e], d = rcv[e];
    float hsc = bf2f(hsbf[(size_t)s * F_DIM + c]);
    const float* gar = gA1 + (size_t)d * 2048 + c;
    #pragma unroll
    for (int k = 0; k < 16; ++k) part[k][c] = gar[k * F_DIM] * hsc;
    __syncthreads();
    if (c < 16) {
        float sacc = 0.f;
        for (int i = 0; i < F_DIM; ++i) sacc += part[c][i];
        dots[c] = sacc;
    }
    __syncthreads();
    const float* Ye = Y + (size_t)e * 16;
    const float* Re = R1 + (size_t)e * 4;
    const float inv = 1.f / AVGF;
    if (c < 16) gY[(size_t)e * 16 + c] += Re[LMAP_c[c]] * dots[c] * inv;
    if (c < 4) {
        float v = 0.f;
        for (int k = 0; k < 16; ++k) if (LMAP_c[k] == c) v += Ye[k] * dots[k];
        grloc[c] = v * inv;
    }
    __syncthreads();
    if (c < 32) gR1bf[(size_t)e * 32 + c] = (c < 4) ? f2bf(grloc[c]) : (u16b)0;
}

// ---------------- forces ----------------
__global__ void k_forces(const float* __restrict__ rbuf, const float* __restrict__ ubuf,
                         const float* __restrict__ gY, const float* __restrict__ gef,
                         const int* __restrict__ snd, const int* __restrict__ rcv,
                         float* __restrict__ forces, int E)
{
    int e = blockIdx.x * blockDim.x + threadIdx.x;
    if (e >= E) return;
    float r = rbuf[e];
    float x = ubuf[e * 3 + 0], y = ubuf[e * 3 + 1], z = ubuf[e * 3 + 2];
    const float* gy = gY + (size_t)e * 16;

    const float s3  = 1.7320508075688772f;
    const float s5  = 2.2360679774997896f;
    const float s15 = 3.872983346207417f;
    const float c9  = 2.0916500663351889f;
    const float c10 = 10.246950765959598f;
    const float c11 = 1.6201851746019651f;
    const float c12 = 1.3228756555322954f;
    const float c14 = 5.123475382979799f;

    float gux = 0.f, guy = 0.f, guz = 0.f;
    gux += gy[1] * s3;
    guy += gy[2] * s3;
    guz += gy[3] * s3;
    gux += gy[4] * s15 * y;            guy += gy[4] * s15 * x;
    guy += gy[5] * s15 * z;            guz += gy[5] * s15 * y;
    guz += gy[6] * 3.f * s5 * z;
    gux += gy[7] * s15 * z;            guz += gy[7] * s15 * x;
    gux += gy[8] * s15 * x;            guy -= gy[8] * s15 * y;
    gux += gy[9] * c9 * 6.f * x * y;   guy += gy[9] * c9 * 3.f * (x * x - y * y);
    gux += gy[10] * c10 * y * z;       guy += gy[10] * c10 * x * z;  guz += gy[10] * c10 * x * y;
    guy += gy[11] * c11 * (5.f * z * z - 1.f);  guz += gy[11] * c11 * 10.f * y * z;
    guz += gy[12] * c12 * (15.f * z * z - 3.f);
    gux += gy[13] * c11 * (5.f * z * z - 1.f);  guz += gy[13] * c11 * 10.f * x * z;
    gux += gy[14] * c14 * 2.f * x * z;  guy -= gy[14] * c14 * 2.f * y * z;  guz += gy[14] * c14 * (x * x - y * y);
    gux += gy[15] * c9 * 3.f * (x * x - y * y); guy -= gy[15] * c9 * 6.f * x * y;

    float u = r / RMAXF;
    float cut = 0.f, dcut = 0.f;
    if (u < 1.f) {
        float u2 = u * u;
        float u4 = u2 * u2;
        float u5 = u4 * u;
        cut = 1.f - 21.f * u5 + 35.f * u5 * u - 15.f * u5 * u2;
        float om = 1.f - u;
        dcut = -(105.f / RMAXF) * u4 * om * om;
    }
    float s2r = sqrtf(2.f / RMAXF);
    float sr = 0.f;
    #pragma unroll
    for (int n = 1; n <= NBES; ++n) {
        float w = (float)n * PI_F / RMAXF;
        float sn = __sinf(w * r), cs = __cosf(w * r);
        float b   = s2r * sn / r;
        float db  = s2r * (w * cs / r - sn / (r * r));
        float def = db * cut + b * dcut;
        sr += gef[(size_t)e * NBES + (n - 1)] * def;
    }
    float gdu = gux * x + guy * y + guz * z;
    float gvx = (gux - gdu * x) / r + sr * x;
    float gvy = (guy - gdu * y) / r + sr * y;
    float gvz = (guz - gdu * z) / r + sr * z;
    int s = snd[e], d = rcv[e];
    atomicAdd(&forces[s * 3 + 0],  gvx);
    atomicAdd(&forces[s * 3 + 1],  gvy);
    atomicAdd(&forces[s * 3 + 2],  gvz);
    atomicAdd(&forces[d * 3 + 0], -gvx);
    atomicAdd(&forces[d * 3 + 1], -gvy);
    atomicAdd(&forces[d * 3 + 2], -gvz);
}

// =====================================================================
// Host side
// =====================================================================
enum {
    I_POS, I_ATTR, I_CHG, I_AE, I_WEMB, I_WUP1,
    I_R10, I_R11, I_R12, I_R13,
    I_WMIX1, I_WP01, I_WP11, I_WOUT10, I_WOUT11, I_WE1, I_WD1,
    I_WUP2, I_WUP2V,
    I_R20, I_R21, I_R22, I_R23,
    I_WMIX2, I_WSC2, I_WP02, I_WP12, I_WOUT20, I_WOUT21,
    I_WMLPS, I_WMLPV, I_WE2, I_WD2,
    I_EIDX, I_SHIFTS, I_BATCH, I_COUNT
};
static const int MAP_INS[I_COUNT] = {
    0, 1, 2, 3, 4, 5, 6, 7, 8, 9, 10, 11, 12, 13, 14, 15, 16,
    17, 18, 19, 20, 21, 22, 23, 24, 25, 26, 27, 28, 29, 30, 31, 32, 33, 34, 35
};
static const int MAP_ALPHA[I_COUNT] = {
    34, 3, 1, 17, 4, 14, 18, 19, 20, 21, 5, 30, 32, 9, 10, 28, 26,
    15, 16, 22, 23, 24, 25, 6, 13, 31, 33, 11, 12, 7, 8, 29, 27, 2, 35, 0
};

static inline char* ws_take(char*& p, size_t bytes) {
    char* r = p;
    p += (bytes + 255) & ~(size_t)255;
    return r;
}

extern "C" void kernel_launch(void* const* d_in, const int* in_sizes, int n_in,
                              void* d_out, int out_size, void* d_ws, size_t ws_size,
                              hipStream_t stream)
{
    const int* map = (n_in > 0 && in_sizes[0] == 12000) ? MAP_INS : MAP_ALPHA;
    const float* IN_F[I_COUNT];
    for (int i = 0; i < I_COUNT; ++i) IN_F[i] = (const float*)d_in[map[i]];

    const float* pos     = IN_F[I_POS];
    const float* attrs   = IN_F[I_ATTR];
    const float* charges = IN_F[I_CHG];
    const int*   eidx    = (const int*)d_in[map[I_EIDX]];
    const float* shifts  = IN_F[I_SHIFTS];
    const int*   batch   = (const int*)d_in[map[I_BATCH]];

    const int N = in_sizes[map[I_POS]] / 3;
    const int E = in_sizes[map[I_EIDX]] / 2;
    const int* snd = eidx;
    const int* rcv = eidx + E;

    float* out    = (float*)d_out;
    float* te     = out;
    float* ne0    = out + G_DIM;
    float* forces = ne0 + N;
    float* td     = forces + 3 * N;
    float* ad     = td + 3 * G_DIM;
    (void)out_size; (void)ws_size;

    // ---------------- workspace ----------------
    char* w = (char*)d_ws;
    float* rbuf  = (float*)ws_take(w, (size_t)E * 4);
    float* ubuf  = (float*)ws_take(w, (size_t)E * 12);
    float* Y     = (float*)ws_take(w, (size_t)E * 16 * 4);
    float* gYb   = (float*)ws_take(w, (size_t)E * 16 * 4);
    u16b* efbf   = (u16b*)ws_take(w, (size_t)E * 32 * 2);
    u16b* t1     = (u16b*)ws_take(w, (size_t)E * 64 * 2);
    u16b* t2     = (u16b*)ws_take(w, (size_t)E * 64 * 2);
    u16b* z11    = (u16b*)ws_take(w, (size_t)E * 64 * 2);
    u16b* z12    = (u16b*)ws_take(w, (size_t)E * 64 * 2);
    u16b* z13    = (u16b*)ws_take(w, (size_t)E * 64 * 2);
    u16b* z21    = (u16b*)ws_take(w, (size_t)E * 64 * 2);
    u16b* z22    = (u16b*)ws_take(w, (size_t)E * 64 * 2);
    u16b* z23    = (u16b*)ws_take(w, (size_t)E * 64 * 2);
    float* R1f   = (float*)ws_take(w, (size_t)E * 4 * 4);
    float* R2f   = (float*)ws_take(w, (size_t)E * 5 * 4);
    u16b* gR1bf  = (u16b*)ws_take(w, (size_t)E * 32 * 2);
    u16b* gR2bf  = (u16b*)ws_take(w, (size_t)E * 32 * 2);
    float* geff  = (float*)ws_take(w, (size_t)E * NBES * 4);
    int*  typ    = (int*)ws_take(w, (size_t)N * 4);
    u16b* hbf    = (u16b*)ws_take(w, (size_t)N * F_DIM * 2);
    u16b* hsbf   = (u16b*)ws_take(w, (size_t)N * F_DIM * 2);
    float* Araw  = (float*)ws_take(w, (size_t)N * 2048 * 4);  // scatter, Amix2, gA2, gA1 (sequentially)
    u16b* Abf    = (u16b*)ws_take(w, (size_t)N * 2048 * 2);
    float* Amix1 = (float*)ws_take(w, (size_t)N * 2048 * 4);
    float* gbig  = (float*)ws_take(w, (size_t)N * 2048 * 4);
    u16b* h0in   = (u16b*)ws_take(w, (size_t)N * F_DIM * 2);
    u16b* h1in   = (u16b*)ws_take(w, (size_t)N * 3 * F_DIM * 2);
    float* h0f   = (float*)ws_take(w, (size_t)N * F_DIM * 4);
    u16b* h0bf   = (u16b*)ws_take(w, (size_t)N * F_DIM * 2);
    float* h1f   = (float*)ws_take(w, (size_t)N * 3 * F_DIM * 4);
    u16b* h1bf   = (u16b*)ws_take(w, (size_t)N * 3 * F_DIM * 2);
    u16b* hs0bf  = (u16b*)ws_take(w, (size_t)N * F_DIM * 2);
    u16b* hvbf   = (u16b*)ws_take(w, (size_t)N * 3 * F_DIM * 2);
    float* B0f   = (float*)ws_take(w, (size_t)N * F_DIM * 4);
    u16b* B0bf   = (u16b*)ws_take(w, (size_t)N * F_DIM * 2);
    float* B1f   = (float*)ws_take(w, (size_t)N * 3 * F_DIM * 4);
    u16b* B1bf   = (u16b*)ws_take(w, (size_t)N * 3 * F_DIM * 2);
    float* sbuf  = (float*)ws_take(w, (size_t)N * 32 * 4);
    float* vbuf  = (float*)ws_take(w, (size_t)N * 48 * 4);
    float* d1buf = (float*)ws_take(w, (size_t)N * 3 * 4);
    u16b* dsbf   = (u16b*)ws_take(w, (size_t)N * 32 * 2);
    float* gB0f  = (float*)ws_take(w, (size_t)N * F_DIM * 4);
    u16b* gB0bf  = (u16b*)ws_take(w, (size_t)N * F_DIM * 2);
    float* gb0in = (float*)ws_take(w, (size_t)N * F_DIM * 4);
    float* ghs0f = (float*)ws_take(w, (size_t)N * F_DIM * 4);
    u16b* ghs0bf = (u16b*)ws_take(w, (size_t)N * F_DIM * 2);
    float* ghvf  = (float*)ws_take(w, (size_t)N * 3 * F_DIM * 4);
    u16b* ghvbf  = (u16b*)ws_take(w, (size_t)N * 3 * F_DIM * 2);
    float* gh0f  = (float*)ws_take(w, (size_t)N * F_DIM * 4);
    u16b* gh0bf  = (u16b*)ws_take(w, (size_t)N * F_DIM * 2);
    u16b* gh1bf  = (u16b*)ws_take(w, (size_t)N * 3 * F_DIM * 2);
    float* gh0i  = (float*)ws_take(w, (size_t)N * F_DIM * 4);
    float* gh1i  = (float*)ws_take(w, (size_t)N * 3 * F_DIM * 4);

    // packed weights (forward)
    u16b* P_R10   = (u16b*)ws_take(w, 2048 * 2);
    u16b* P_R11   = (u16b*)ws_take(w, 4096 * 2);
    u16b* P_R12   = (u16b*)ws_take(w, 4096 * 2);
    u16b* P_R13   = (u16b*)ws_take(w, 1024 * 2);
    u16b* P_R20   = (u16b*)ws_take(w, 2048 * 2);
    u16b* P_R21   = (u16b*)ws_take(w, 4096 * 2);
    u16b* P_R22   = (u16b*)ws_take(w, 4096 * 2);
    u16b* P_R23   = (u16b*)ws_take(w, 1024 * 2);
    u16b* P_WUP1  = (u16b*)ws_take(w, 16384 * 2);
    u16b* P_MIX1  = (u16b*)ws_take(w, 65536 * 2);
    u16b* P_MIX2  = (u16b*)ws_take(w, 65536 * 2);
    u16b* P_O10   = (u16b*)ws_take(w, 16384 * 2);
    u16b* P_O11   = (u16b*)ws_take(w, 16384 * 2);
    u16b* P_WUP2  = (u16b*)ws_take(w, 16384 * 2);
    u16b* P_WUP2V = (u16b*)ws_take(w, 16384 * 2);
    u16b* P_O20   = (u16b*)ws_take(w, 16384 * 2);
    u16b* P_O21   = (u16b*)ws_take(w, 16384 * 2);
    u16b* P_MLPS  = (u16b*)ws_take(w, 4096 * 2);
    u16b* P_MLPV  = (u16b*)ws_take(w, 2048 * 2);
    // packed weights (transposed, backward)
    u16b* PT_MLPS = (u16b*)ws_take(w, 4096 * 2);
    u16b* PT_O20  = (u16b*)ws_take(w, 16384 * 2);
    u16b* PT_O10  = (u16b*)ws_take(w, 16384 * 2);
    u16b* PT_O11  = (u16b*)ws_take(w, 16384 * 2);
    u16b* PT_UP2  = (u16b*)ws_take(w, 16384 * 2);
    u16b* PT_UP2V = (u16b*)ws_take(w, 16384 * 2);
    u16b* PT_MIX1 = (u16b*)ws_take(w, 65536 * 2);
    u16b* PT_MIX2 = (u16b*)ws_take(w, 65536 * 2);
    u16b* PT_R10  = (u16b*)ws_take(w, 1024 * 2);
    u16b* PT_R11  = (u16b*)ws_take(w, 4096 * 2);
    u16b* PT_R12  = (u16b*)ws_take(w, 4096 * 2);
    u16b* PT_R13  = (u16b*)ws_take(w, 2048 * 2);
    u16b* PT_R20  = (u16b*)ws_take(w, 1024 * 2);
    u16b* PT_R21  = (u16b*)ws_take(w, 4096 * 2);
    u16b* PT_R22  = (u16b*)ws_take(w, 4096 * 2);
    u16b* PT_R23  = (u16b*)ws_take(w, 2048 * 2);

    auto pack = [&](const float* W, u16b* dst, int Kvalid, int Kpad, int Nc, int rs, int cs) {
        int kblocks = Kpad >> 5;
        int tilesN = (Nc + 15) >> 4;
        int total = kblocks * tilesN * 512;
        k_pack_b<<<(total + 255) / 256, 256, 0, stream>>>(W, dst, Kvalid, Nc, rs, cs, kblocks, tilesN);
    };
    pack(IN_F[I_R10], P_R10, 8, 32, 64, 64, 1);
    pack(IN_F[I_R11], P_R11, 64, 64, 64, 64, 1);
    pack(IN_F[I_R12], P_R12, 64, 64, 64, 64, 1);
    pack(IN_F[I_R13], P_R13, 64, 64, 4, 4, 1);
    pack(IN_F[I_R20], P_R20, 8, 32, 64, 64, 1);
    pack(IN_F[I_R21], P_R21, 64, 64, 64, 64, 1);
    pack(IN_F[I_R22], P_R22, 64, 64, 64, 64, 1);
    pack(IN_F[I_R23], P_R23, 64, 64, 5, 5, 1);
    pack(IN_F[I_WUP1], P_WUP1, 128, 128, 128, 128, 1);
    for (int l = 0; l < 4; ++l) {
        pack(IN_F[I_WMIX1] + (size_t)l * 16384, P_MIX1 + (size_t)l * 16384, 128, 128, 128, 128, 1);
        pack(IN_F[I_WMIX2] + (size_t)l * 16384, P_MIX2 + (size_t)l * 16384, 128, 128, 128, 128, 1);
        pack(IN_F[I_WMIX1] + (size_t)l * 16384, PT_MIX1 + (size_t)l * 16384, 128, 128, 128, 1, 128);
        pack(IN_F[I_WMIX2] + (size_t)l * 16384, PT_MIX2 + (size_t)l * 16384, 128, 128, 128, 1, 128);
    }
    pack(IN_F[I_WOUT10], P_O10, 128, 128, 128, 128, 1);
    pack(IN_F[I_WOUT11], P_O11, 128, 128, 128, 128, 1);
    pack(IN_F[I_WUP2],  P_WUP2, 128, 128, 128, 128, 1);
    pack(IN_F[I_WUP2V], P_WUP2V, 128, 128, 128, 128, 1);
    pack(IN_F[I_WOUT20], P_O20, 128, 128, 128, 128, 1);
    pack(IN_F[I_WOUT21], P_O21, 128, 128, 128, 128, 1);
    pack(IN_F[I_WMLPS], P_MLPS, 128, 128, 32, 32, 1);
    pack(IN_F[I_WMLPV], P_MLPV, 128, 128, 16, 16, 1);
    pack(IN_F[I_WMLPS], PT_MLPS, 32, 32, 128, 1, 32);
    pack(IN_F[I_WOUT20], PT_O20, 128, 128, 128, 1, 128);
    pack(IN_F[I_WOUT10], PT_O10, 128, 128, 128, 1, 128);
    pack(IN_F[I_WOUT11], PT_O11, 128, 128, 128, 1, 128);
    pack(IN_F[I_WUP2],  PT_UP2, 128, 128, 128, 1, 128);
    pack(IN_F[I_WUP2V], PT_UP2V, 128, 128, 128, 1, 128);
    pack(IN_F[I_R10], PT_R10, 64, 64, 8, 1, 64);
    pack(IN_F[I_R11], PT_R11, 64, 64, 64, 1, 64);
    pack(IN_F[I_R12], PT_R12, 64, 64, 64, 1, 64);
    pack(IN_F[I_R13], PT_R13, 4, 32, 64, 1, 4);
    pack(IN_F[I_R20], PT_R20, 64, 64, 8, 1, 64);
    pack(IN_F[I_R21], PT_R21, 64, 64, 64, 1, 64);
    pack(IN_F[I_R22], PT_R22, 64, 64, 64, 1, 64);
    pack(IN_F[I_R23], PT_R23, 5, 32, 64, 1, 5);

    auto gemm = [&](const u16b* A, const u16b* Bp, float* C, u16b* Cbf,
                    u16b* Zpre, const u16b* Zin,
                    int M, int K, int Nc, int lda, int ldc, int act, int accum) {
        int tilesN = (Nc + 15) / 16;
        int tilesM = M / 16;
        if (tilesN % 4 == 0) {
            int tiles = tilesM * (tilesN / 4);
            k_gemm_bf16_t<4><<<(tiles + 7) / 8, 256, 0, stream>>>(
                A, Bp, C, Cbf, Zpre, Zin, M, K, Nc, lda, ldc, 1.f, act, accum);
        } else if (tilesN % 2 == 0) {
            int tiles = tilesM * (tilesN / 2);
            k_gemm_bf16_t<2><<<(tiles + 7) / 8, 256, 0, stream>>>(
                A, Bp, C, Cbf, Zpre, Zin, M, K, Nc, lda, ldc, 1.f, act, accum);
        } else {
            int tiles = tilesM * tilesN;
            k_gemm_bf16_t<1><<<(tiles + 7) / 8, 256, 0, stream>>>(
                A, Bp, C, Cbf, Zpre, Zin, M, K, Nc, lda, ldc, 1.f, act, accum);
        }
    };
    auto zero = [&](float* p, long n) {
        int blocks = (int)((n + 255) / 256);
        if (blocks > 2048) blocks = 2048;
        k_zero_f32<<<blocks, 256, 0, stream>>>(p, n);
    };
    auto tobf = [&](const float* src, u16b* dst, long n, float scale) {
        int blocks = (int)((n + 255) / 256);
        if (blocks > 2048) blocks = 2048;
        if (scale == 1.f) k_f2bf_arr<<<blocks, 256, 0, stream>>>(src, dst, n);
        else              k_f2bf_scale<<<blocks, 256, 0, stream>>>(src, dst, n, scale);
    };

    // =================== FORWARD ===================
    zero(te, G_DIM);
    zero(td, 3 * G_DIM);
    zero(forces, (long)3 * N);

    k_edges<<<(E + 255) / 256, 256, 0, stream>>>(pos, shifts, snd, rcv, rbuf, ubuf, Y, efbf, E);
    k_embed<<<N, F_DIM, 0, stream>>>(attrs, IN_F[I_AE], IN_F[I_WEMB], batch, typ, hbf, ne0, te, N);

    gemm(efbf, P_R10, nullptr, t1, z11, nullptr, E, 32, 64, 32, 64, 1, 0);
    gemm(t1, P_R11, nullptr, t2, z12, nullptr, E, 64, 64, 64, 64, 1, 0);
    gemm(t2, P_R12, nullptr, t1, z13, nullptr, E, 64, 64, 64, 64, 1, 0);
    gemm(t1, P_R13, R1f, nullptr, nullptr, nullptr, E, 64, 4, 64, 4, 0, 0);
    gemm(efbf, P_R20, nullptr, t1, z21, nullptr, E, 32, 64, 32, 64, 1, 0);
    gemm(t1, P_R21, nullptr, t2, z22, nullptr, E, 64, 64, 64, 64, 1, 0);
    gemm(t2, P_R22, nullptr, t1, z23, nullptr, E, 64, 64, 64, 64, 1, 0);
    gemm(t1, P_R23, R2f, nullptr, nullptr, nullptr, E, 64, 5, 64, 5, 0, 0);

    gemm(hbf, P_WUP1, nullptr, hsbf, nullptr, nullptr, N, 128, 128, 128, 128, 0, 0);

    zero(Araw, (long)N * 2048);
    k_msg1<<<E, F_DIM, 0, stream>>>(Y, R1f, hsbf, snd, rcv, Araw);
    tobf(Araw, Abf, (long)N * 2048, 1.f / AVGF);
    for (int k = 0; k < 16; ++k)
        gemm(Abf + k * F_DIM, P_MIX1 + (size_t)LMAP_h[k] * 16384,
             Amix1 + k * F_DIM, nullptr, nullptr, nullptr, N, 128, 128, 2048, 2048, 0, 0);
    k_nonlin<<<N, F_DIM, 0, stream>>>(Amix1, IN_F[I_WP01], IN_F[I_WP11], h0in, h1in);
    gemm(h0in, P_O10, h0f, h0bf, nullptr, nullptr, N, 128, 128, 128, 128, 0, 0);
    gemm(h1in, P_O11, h1f, h1bf, nullptr, nullptr, 3 * N, 128, 128, 128, 128, 0, 0);
    k_e1d1<<<(N + 127) / 128, 128, 0, stream>>>(h0f, h1f, IN_F[I_WE1], IN_F[I_WD1], batch, te, d1buf, N);
    gemm(h0bf, P_WUP2, nullptr, hs0bf, nullptr, nullptr, N, 128, 128, 128, 128, 0, 0);
    gemm(h1bf, P_WUP2V, nullptr, hvbf, nullptr, nullptr, 3 * N, 128, 128, 128, 128, 0, 0);

    zero(Araw, (long)N * 2048);
    k_msg2<<<E, F_DIM, 0, stream>>>(Y, R2f, hs0bf, hvbf, snd, rcv, Araw);
    tobf(Araw, Abf, (long)N * 2048, 1.f / AVGF);
    for (int k = 0; k < 16; ++k)
        gemm(Abf + k * F_DIM, P_MIX2 + (size_t)LMAP_h[k] * 16384,
             Araw + k * F_DIM, nullptr, nullptr, nullptr, N, 128, 128, 2048, 2048, 0, 0);
    k_nonlin<<<N, F_DIM, 0, stream>>>(Araw, IN_F[I_WP02], IN_F[I_WP12], h0in, h1in);
    gemm(h0in, P_O20, B0f, nullptr, nullptr, nullptr, N, 128, 128, 128, 128, 0, 0);
    gemm(h1in, P_O21, B1f, nullptr, nullptr, nullptr, 3 * N, 128, 128, 128, 128, 0, 0);
    k_sc_add<<<N, F_DIM, 0, stream>>>(h0f, h1f, typ, IN_F[I_WSC2], B0f, B1f, B0bf, B1bf);
    gemm(B0bf, P_MLPS, sbuf, nullptr, nullptr, nullptr, N, 128, 32, 128, 32, 0, 0);
    gemm(B1bf, P_MLPV, vbuf, nullptr, nullptr, nullptr, 3 * N, 128, 16, 128, 16, 0, 0);
    k_readout<<<(N + 255) / 256, 256, 0, stream>>>(sbuf, vbuf, d1buf, IN_F[I_WE2], IN_F[I_WD2],
                                                   batch, charges, pos, te, td, ad, N);

    // =================== BACKWARD (energy -> positions) ===================
    k_ds<<<(N * 32 + 255) / 256, 256, 0, stream>>>(sbuf, IN_F[I_WE2], dsbf, N);
    gemm(dsbf, PT_MLPS, gB0f, gB0bf, nullptr, nullptr, N, 32, 128, 32, 128, 0, 0);
    gemm(gB0bf, PT_O20, gb0in, nullptr, nullptr, nullptr, N, 128, 128, 128, 128, 0, 0);
    k_nonlin_bwd<<<N, F_DIM, 0, stream>>>(Araw, gb0in, nullptr, IN_F[I_WP02], IN_F[I_WP12], gbig);
    tobf(gbig, Abf, (long)N * 2048, 1.f);
    for (int k = 0; k < 16; ++k)
        gemm(Abf + k * F_DIM, PT_MIX2 + (size_t)LMAP_h[k] * 16384,
             Araw + k * F_DIM, nullptr, nullptr, nullptr, N, 128, 128, 2048, 2048, 0, 0);
    zero(ghs0f, (long)N * F_DIM);
    zero(ghvf, (long)N * 3 * F_DIM);
    k_edge_bwd2<<<E, F_DIM, 0, stream>>>(Y, R2f, hs0bf, hvbf, Araw, snd, rcv,
                                         gYb, gR2bf, ghs0f, ghvf);
    tobf(ghs0f, ghs0bf, (long)N * F_DIM, 1.f);
    tobf(ghvf, ghvbf, (long)N * 3 * F_DIM, 1.f);
    k_sc_bwd<<<N, F_DIM, 0, stream>>>(gB0f, typ, IN_F[I_WSC2], IN_F[I_WE1], gh0f);
    gemm(ghs0bf, PT_UP2, gh0f, gh0bf, nullptr, nullptr, N, 128, 128, 128, 128, 0, 1);
    gemm(ghvbf, PT_UP2V, nullptr, gh1bf, nullptr, nullptr, 3 * N, 128, 128, 128, 128, 0, 0);
    gemm(gh0bf, PT_O10, gh0i, nullptr, nullptr, nullptr, N, 128, 128, 128, 128, 0, 0);
    gemm(gh1bf, PT_O11, gh1i, nullptr, nullptr, nullptr, 3 * N, 128, 128, 128, 128, 0, 0);
    k_nonlin_bwd<<<N, F_DIM, 0, stream>>>(Amix1, gh0i, gh1i, IN_F[I_WP01], IN_F[I_WP11], gbig);
    tobf(gbig, Abf, (long)N * 2048, 1.f);
    for (int k = 0; k < 16; ++k)
        gemm(Abf + k * F_DIM, PT_MIX1 + (size_t)LMAP_h[k] * 16384,
             Araw + k * F_DIM, nullptr, nullptr, nullptr, N, 128, 128, 2048, 2048, 0, 0);
    k_edge_bwd1<<<E, F_DIM, 0, stream>>>(Y, R1f, hsbf, Araw, snd, rcv, gYb, gR1bf);

    gemm(gR1bf, PT_R13, nullptr, t1, nullptr, z13, E, 32, 64, 32, 64, 2, 0);
    gemm(t1, PT_R12, nullptr, t2, nullptr, z12, E, 64, 64, 64, 64, 2, 0);
    gemm(t2, PT_R11, nullptr, t1, nullptr, z11, E, 64, 64, 64, 64, 2, 0);
    gemm(t1, PT_R10, geff, nullptr, nullptr, nullptr, E, 64, 8, 64, 8, 0, 0);
    gemm(gR2bf, PT_R23, nullptr, t1, nullptr, z23, E, 32, 64, 32, 64, 2, 0);
    gemm(t1, PT_R22, nullptr, t2, nullptr, z22, E, 64, 64, 64, 64, 2, 0);
    gemm(t2, PT_R21, nullptr, t1, nullptr, z21, E, 64, 64, 64, 64, 2, 0);
    gemm(t1, PT_R20, geff, nullptr, nullptr, nullptr, E, 64, 8, 64, 8, 0, 1);

    k_forces<<<(E + 255) / 256, 256, 0, stream>>>(rbuf, ubuf, gYb, geff, snd, rcv, forces, E);
}
